// NeuralCausalDiscovery_76897094468060
// MI455X (gfx1250) — matrix operand order repaired and checked by reading, not compiled, not run on hardware
//
#include <hip/hip_runtime.h>
#include <hip/hip_bf16.h>

typedef __attribute__((ext_vector_type(16))) _Float16 v16h;
typedef __attribute__((ext_vector_type(8)))  float    v8f;

#define NVARS 28
#define HID   64
#define BATCH 32768
#define NCHUNK 8
#define ROWS_PER_CHUNK (BATCH / NCHUNK)   // 4096
#define TILE_M 128
#define NW 8                               // waves per 256-thread block
#define GRAM_BLOCKS 64
#define ROWS_PER_GBLOCK (BATCH / GRAM_BLOCKS) // 512
#define BN_EPS 1e-5f

// ---------------- workspace layout (bytes) ----------------
static constexpr size_t OFF_WMAT  = 0;                                  // 784 f32
static constexpr size_t OFF_XH    = 4096;                               // BATCH*32 f16
static constexpr size_t OFF_W1H   = OFF_XH   + (size_t)BATCH * 32 * 2;  // 28*64*32 f16
static constexpr size_t OFF_W2H   = OFF_W1H  + (size_t)NVARS * HID * 32 * 2; // 28*64*64 f16
static constexpr size_t OFF_GRAMP = OFF_W2H  + (size_t)NVARS * HID * HID * 2; // 64*784 f32
static constexpr size_t OFF_XSUMP = OFF_GRAMP + (size_t)GRAM_BLOCKS * 784 * 4; // 64*28 f32
static constexpr size_t OFF_SC1   = OFF_XSUMP + (size_t)GRAM_BLOCKS * NVARS * 4;
static constexpr size_t OFF_SH1   = OFF_SC1 + (size_t)NVARS * HID * 4;
static constexpr size_t OFF_SC2   = OFF_SH1 + (size_t)NVARS * HID * 4;
static constexpr size_t OFF_SH2   = OFF_SC2 + (size_t)NVARS * HID * 4;
static constexpr size_t OFF_SPART = OFF_SH2 + (size_t)NVARS * HID * 4;  // 28*8*8*64 f32
static constexpr size_t OFF_QPART = OFF_SPART + (size_t)NVARS * NCHUNK * NW * HID * 4;
static constexpr size_t WS_NEEDED = OFF_QPART + (size_t)NVARS * NCHUNK * NW * HID * 4;

__device__ __forceinline__ float dev_gelu(float x) {
    return 0.5f * x * (1.0f + erff(x * 0.70710678118654752f));
}

__device__ __forceinline__ v8f wmma_f16(v16h a, v16h b, v8f c) {
    return __builtin_amdgcn_wmma_f32_16x16x32_f16(false, a, false, b, (short)0, c, false, false);
}

union FragU { v16h v; uint4 q[2]; };

// A fragment: 16xK tile, row-major f16, leading dim ld (halves).
// lane: M = lane&15, half = lane>>4; elems 0..7 = K[kstep+8h .. +7], 8..15 = K[kstep+16+8h .. +7]
__device__ __forceinline__ v16h load_a_frag(const _Float16* base, int ld, int kstep) {
    int lane = threadIdx.x & 31;
    int m = lane & 15, hf = (lane >> 4) & 1;
    const _Float16* p = base + (size_t)m * ld + kstep + 8 * hf;
    FragU u;
    u.q[0] = *(const uint4*)p;
    u.q[1] = *(const uint4*)(p + 16);
    return u.v;
}

// B fragment: weights stored row-major [Ncol][K], leading dim ld.
// lane: N = 16*coltile + (lane&15); elems 0..7 = K[kstep+8h .. +7], 8..15 = K[kstep+16+8h .. +7]
__device__ __forceinline__ v16h load_b_frag(const _Float16* w, int ld, int coltile, int kstep) {
    int lane = threadIdx.x & 31;
    int nn = lane & 15, hf = (lane >> 4) & 1;
    const _Float16* p = w + (size_t)(coltile * 16 + nn) * ld + kstep + 8 * hf;
    FragU u;
    u.q[0] = *(const uint4*)p;
    u.q[1] = *(const uint4*)(p + 16);
    return u.v;
}

// ---------------- K1a: adjacency W = sigmoid(Wl)*(1-I) ----------------
__global__ void k_wmat(const float* __restrict__ Wlog, float* __restrict__ wmat,
                       float* __restrict__ outW) {
    for (int i = threadIdx.x; i < NVARS * NVARS; i += blockDim.x) {
        int r = i / NVARS, c = i % NVARS;
        float w = (r == c) ? 0.0f : 1.0f / (1.0f + expf(-Wlog[i]));
        wmat[i] = w;
        outW[i] = w;
    }
}

// ---------------- K1b: f16 conversions (X padded to 32, W1*mask, W2) ----------------
__global__ void k_convert(const float* __restrict__ X, const float* __restrict__ W1,
                          const float* __restrict__ W2, const float* __restrict__ wmat,
                          _Float16* __restrict__ Xh, _Float16* __restrict__ W1h,
                          _Float16* __restrict__ W2h) {
    const int totalX = BATCH * 32;
    const int totalW1 = NVARS * HID * 32;
    const int totalW2 = NVARS * HID * HID;
    for (int i = blockIdx.x * blockDim.x + threadIdx.x; i < totalX + totalW1 + totalW2;
         i += gridDim.x * blockDim.x) {
        if (i < totalX) {
            int b = i >> 5, d = i & 31;
            Xh[i] = (d < NVARS) ? (_Float16)X[b * NVARS + d] : (_Float16)0.0f;
        } else if (i < totalX + totalW1) {
            int j = i - totalX;
            int d = j & 31, nh = j >> 5, n = nh / HID;
            W1h[j] = (d < NVARS) ? (_Float16)(W1[nh * NVARS + d] * wmat[n * NVARS + d])
                                 : (_Float16)0.0f;
        } else {
            int j = i - totalX - totalW1;
            W2h[j] = (_Float16)W2[j];
        }
    }
}

// ---------------- K2: batch Gram matrix + column sums of X (partial per block) ----------------
__global__ void k_gram(const float* __restrict__ X, float* __restrict__ gramPart,
                       float* __restrict__ xsumPart) {
    __shared__ float tileX[64 * NVARS];
    int tid = threadIdx.x, blk = blockIdx.x;
    int d = tid / NVARS, e = tid % NVARS;   // valid when tid < 784
    float acc = 0.0f, accs = 0.0f;
    for (int it = 0; it < ROWS_PER_GBLOCK / 64; ++it) {
        int rbase = blk * ROWS_PER_GBLOCK + it * 64;
        for (int i = tid; i < 64 * NVARS; i += blockDim.x)
            tileX[i] = X[(size_t)rbase * NVARS + i];
        __syncthreads();
        if (tid < NVARS * NVARS) {
            for (int r = 0; r < 64; ++r) acc += tileX[r * NVARS + d] * tileX[r * NVARS + e];
        }
        if (tid >= 784 && tid < 784 + NVARS) {
            int ee = tid - 784;
            for (int r = 0; r < 64; ++r) accs += tileX[r * NVARS + ee];
        }
        __syncthreads();
    }
    if (tid < 784) gramPart[blk * 784 + tid] = acc;
    if (tid >= 784 && tid < 784 + NVARS) xsumPart[blk * NVARS + (tid - 784)] = accs;
}

// ---------------- K3: analytic BN1 stats -> scale1/shift1 (b1 cancels under BN) ----------------
__global__ void k_bn1(const float* __restrict__ gramPart, const float* __restrict__ xsumPart,
                      const _Float16* __restrict__ W1h, const float* __restrict__ g1,
                      const float* __restrict__ be1, float* __restrict__ scale1,
                      float* __restrict__ shift1) {
    int n = blockIdx.x, h = threadIdx.x;   // 64 threads
    __shared__ float xbar[NVARS];
    __shared__ float C[NVARS * NVARS];
    for (int i = h; i < NVARS; i += 64) {
        float s = 0.0f;
        for (int b = 0; b < GRAM_BLOCKS; ++b) s += xsumPart[b * NVARS + i];
        xbar[i] = s / (float)BATCH;
    }
    __syncthreads();
    for (int i = h; i < NVARS * NVARS; i += 64) {
        float s = 0.0f;
        for (int b = 0; b < GRAM_BLOCKS; ++b) s += gramPart[b * 784 + i];
        int dd = i / NVARS, ee = i % NVARS;
        C[i] = s / (float)BATCH - xbar[dd] * xbar[ee];
    }
    __syncthreads();
    float w[NVARS];
    for (int d = 0; d < NVARS; ++d) w[d] = (float)W1h[(n * HID + h) * 32 + d];
    float mu = 0.0f;
    for (int d = 0; d < NVARS; ++d) mu += w[d] * xbar[d];
    float var = 0.0f;
    for (int d = 0; d < NVARS; ++d) {
        float a = 0.0f;
        for (int e = 0; e < NVARS; ++e) a += w[e] * C[d * NVARS + e];
        var += w[d] * a;
    }
    float sc = g1[n * HID + h] * rsqrtf(fmaxf(var, 0.0f) + BN_EPS);
    scale1[n * HID + h] = sc;
    shift1[n * HID + h] = be1[n * HID + h] - mu * sc;
}

// ---------------- fused pass: L1 GEMM -> BN1+GELU -> L2 GEMM -> (stats | BN2+GELU+L3) ----------------
// Weights & BN params live in registers (loop-invariant); only per-wave a1 restage uses LDS.
// No barriers: each wave touches only its own LDS slice; LDS ops are in-order per wave.
template <int PASS>
__global__ __launch_bounds__(256) void k_pass(
    const _Float16* __restrict__ Xh, const _Float16* __restrict__ W1h,
    const _Float16* __restrict__ W2h, const float* __restrict__ scale1,
    const float* __restrict__ shift1, float* __restrict__ sPart, float* __restrict__ qPart,
    const float* __restrict__ scale2, const float* __restrict__ shift2,
    const float* __restrict__ W3, const float* __restrict__ b3, float* __restrict__ Xhat) {
    const int n = blockIdx.y, chunk = blockIdx.x;
    const int tid = threadIdx.x, lane = tid & 31, wave = tid >> 5;
    const int nn = lane & 15, hf = (lane >> 4) & 1;

    __shared__ alignas(16) _Float16 sA[NW][16 * HID];   // per-wave a1 restage

    // ---- hoist weights into registers (stay resident across all tiles) ----
    const _Float16* w1p = W1h + (size_t)n * HID * 32;
    const _Float16* w2p = W2h + (size_t)n * HID * HID;
    v16h bw1[4];
#pragma unroll
    for (int c = 0; c < 4; ++c) bw1[c] = load_b_frag(w1p, 32, c, 0);
    v16h bw2[2][4];
#pragma unroll
    for (int ks = 0; ks < 2; ++ks)
#pragma unroll
        for (int c = 0; c < 4; ++c) bw2[ks][c] = load_b_frag(w2p, HID, c, ks * 32);

    // per-lane BN params (column = c*16+nn is loop-invariant)
    float sc1r[4], sh1r[4], sc2r[4], sh2r[4], w3r[4];
#pragma unroll
    for (int c = 0; c < 4; ++c) {
        int col = n * HID + c * 16 + nn;
        sc1r[c] = scale1[col];
        sh1r[c] = shift1[col];
        if (PASS == 2) {
            sc2r[c] = scale2[col];
            sh2r[c] = shift2[col];
            w3r[c] = W3[col];
        }
    }
    const float b3n = (PASS == 2) ? b3[n] : 0.0f;

    float ssum[4] = {0, 0, 0, 0}, ssq[4] = {0, 0, 0, 0};

    const int ntiles = ROWS_PER_CHUNK / TILE_M;   // 32
    for (int t = 0; t < ntiles; ++t) {
        const int rowbase = chunk * ROWS_PER_CHUNK + t * TILE_M + wave * 16;

        // ---- layer 1: [16x32(K=28 padded)] x [32x64] via 4 WMMA ----
        v16h a1 = load_a_frag(Xh + (size_t)rowbase * 32, 32, 0);
        v8f acc1[4];
#pragma unroll
        for (int c = 0; c < 4; ++c) {
            v8f z = {0, 0, 0, 0, 0, 0, 0, 0};
            acc1[c] = wmma_f16(a1, bw1[c], z);
        }
        // BN1 + GELU, restage a1 tile as f16 row-major [16][64] (wave-private)
#pragma unroll
        for (int c = 0; c < 4; ++c) {
            int col = c * 16 + nn;
#pragma unroll
            for (int r = 0; r < 8; ++r) {
                float v = dev_gelu(sc1r[c] * acc1[c][r] + sh1r[c]);
                sA[wave][(hf * 8 + r) * HID + col] = (_Float16)v;
            }
        }

        // ---- layer 2: [16x64] x [64x64] via 8 WMMA (2 K-steps) ----
        v8f acc2[4];
#pragma unroll
        for (int c = 0; c < 4; ++c) {
            v8f z = {0, 0, 0, 0, 0, 0, 0, 0};
            acc2[c] = z;
        }
#pragma unroll
        for (int ks = 0; ks < 2; ++ks) {
            v16h a2 = load_a_frag(&sA[wave][0], HID, ks * 32);
#pragma unroll
            for (int c = 0; c < 4; ++c) acc2[c] = wmma_f16(a2, bw2[ks][c], acc2[c]);
        }

        if (PASS == 1) {
            // accumulate per-channel sum / sumsq of pre-BN h2
#pragma unroll
            for (int c = 0; c < 4; ++c)
#pragma unroll
                for (int r = 0; r < 8; ++r) {
                    float v = acc2[c][r];
                    ssum[c] += v;
                    ssq[c] += v * v;
                }
        } else {
            // BN2 + GELU + layer-3 dot, reduced across lanes by shuffle butterfly
            float p[8];
#pragma unroll
            for (int r = 0; r < 8; ++r) p[r] = 0.0f;
#pragma unroll
            for (int c = 0; c < 4; ++c)
#pragma unroll
                for (int r = 0; r < 8; ++r)
                    p[r] += dev_gelu(sc2r[c] * acc2[c][r] + sh2r[c]) * w3r[c];
#pragma unroll
            for (int mask = 1; mask <= 8; mask <<= 1)
#pragma unroll
                for (int r = 0; r < 8; ++r) p[r] += __shfl_xor(p[r], mask, 32);
            if (nn == 0) {
#pragma unroll
                for (int r = 0; r < 8; ++r)
                    Xhat[(size_t)(rowbase + hf * 8 + r) * NVARS + n] = p[r] + b3n;
            }
        }
    }

    if (PASS == 1) {
        // combine the two half-wave partials (same column) and store deterministic partials
#pragma unroll
        for (int c = 0; c < 4; ++c) {
            float s = ssum[c] + __shfl_xor(ssum[c], 16, 32);
            float q = ssq[c] + __shfl_xor(ssq[c], 16, 32);
            if (lane < 16) {
                size_t idx = ((((size_t)n * NCHUNK + chunk) * NW + wave) * HID) + c * 16 + lane;
                sPart[idx] = s;
                qPart[idx] = q;
            }
        }
    }
}

// ---------------- K5: finalize BN2 scale/shift from partials ----------------
__global__ void k_bn2(const float* __restrict__ sPart, const float* __restrict__ qPart,
                      const float* __restrict__ g2, const float* __restrict__ be2,
                      float* __restrict__ scale2, float* __restrict__ shift2) {
    int i = blockIdx.x * blockDim.x + threadIdx.x;
    if (i >= NVARS * HID) return;
    int n = i / HID, g = i % HID;
    float s = 0.0f, q = 0.0f;
    for (int c = 0; c < NCHUNK; ++c)
        for (int w = 0; w < NW; ++w) {
            size_t idx = ((((size_t)n * NCHUNK + c) * NW + w) * HID) + g;
            s += sPart[idx];
            q += qPart[idx];
        }
    float mean = s / (float)BATCH;
    float var = q / (float)BATCH - mean * mean;
    float sc = g2[i] * rsqrtf(fmaxf(var, 0.0f) + BN_EPS);
    scale2[i] = sc;
    shift2[i] = be2[i] - mean * sc;
}

extern "C" void kernel_launch(void* const* d_in, const int* in_sizes, int n_in,
                              void* d_out, int out_size, void* d_ws, size_t ws_size,
                              hipStream_t stream) {
    const float* X    = (const float*)d_in[0];
    const float* Wlog = (const float*)d_in[1];
    const float* W1   = (const float*)d_in[2];
    // d_in[3] = b1 : cancels under training-mode BN (constant shift removed by mean)
    const float* g1   = (const float*)d_in[4];
    const float* be1  = (const float*)d_in[5];
    const float* W2   = (const float*)d_in[6];
    // d_in[7] = b2 : cancels under training-mode BN
    const float* g2   = (const float*)d_in[8];
    const float* be2  = (const float*)d_in[9];
    const float* W3   = (const float*)d_in[10];
    const float* b3   = (const float*)d_in[11];

    float* Xhat = (float*)d_out;
    float* outW = Xhat + (size_t)BATCH * NVARS;

    char* ws = (char*)d_ws;
    float*     wmat     = (float*)(ws + OFF_WMAT);
    _Float16*  Xh       = (_Float16*)(ws + OFF_XH);
    _Float16*  W1h      = (_Float16*)(ws + OFF_W1H);
    _Float16*  W2h      = (_Float16*)(ws + OFF_W2H);
    float*     gramPart = (float*)(ws + OFF_GRAMP);
    float*     xsumPart = (float*)(ws + OFF_XSUMP);
    float*     scale1   = (float*)(ws + OFF_SC1);
    float*     shift1   = (float*)(ws + OFF_SH1);
    float*     scale2   = (float*)(ws + OFF_SC2);
    float*     shift2   = (float*)(ws + OFF_SH2);
    float*     sPart    = (float*)(ws + OFF_SPART);
    float*     qPart    = (float*)(ws + OFF_QPART);

    (void)in_sizes; (void)n_in; (void)out_size; (void)ws_size;

    k_wmat<<<1, 256, 0, stream>>>(Wlog, wmat, outW);
    k_convert<<<256, 256, 0, stream>>>(X, W1, W2, wmat, Xh, W1h, W2h);
    k_gram<<<GRAM_BLOCKS, 1024, 0, stream>>>(X, gramPart, xsumPart);
    k_bn1<<<NVARS, 64, 0, stream>>>(gramPart, xsumPart, W1h, g1, be1, scale1, shift1);

    dim3 grid(NCHUNK, NVARS);
    k_pass<1><<<grid, 256, 0, stream>>>(Xh, W1h, W2h, scale1, shift1, sPart, qPart,
                                        nullptr, nullptr, nullptr, nullptr, nullptr);
    k_bn2<<<(NVARS * HID + 255) / 256, 256, 0, stream>>>(sPart, qPart, g2, be2, scale2, shift2);
    k_pass<2><<<grid, 256, 0, stream>>>(Xh, W1h, W2h, scale1, shift1, sPart, qPart,
                                        scale2, shift2, W3, b3, Xhat);
}
